// MultiLabelNonMaxSuppression_71451075936891
// MI455X (gfx1250) — 8-bit matrix operand order repaired and checked by reading, compiled, not run on hardware
//
#include <hip/hip_runtime.h>
#include <hip/hip_bf16.h>

#define N_BOX 8192
#define MASK_WORDS 256

typedef int   v16i __attribute__((ext_vector_type(16)));
typedef float v8f  __attribute__((ext_vector_type(8)));
typedef unsigned long long ull;

__global__ __launch_bounds__(1024) void nms_sort_kernel(
    const float* __restrict__ scores, const float* __restrict__ boxes,
    const int* __restrict__ labels,
    float* __restrict__ sboxes, int* __restrict__ slabels,
    int* __restrict__ order)
{
  __shared__ ull skey[N_BOX];
  const unsigned tid = threadIdx.x;
  #pragma unroll
  for (int s = 0; s < 8; ++s) {
    unsigned i = tid + (unsigned)(s << 10);
    unsigned sb = __float_as_uint(scores[i]);
    skey[i] = ((ull)(~sb) << 32) | (ull)i;
  }
  for (unsigned k = 2; k <= N_BOX; k <<= 1) {
    for (unsigned j = k >> 1; j > 0; j >>= 1) {
      __syncthreads();
      #pragma unroll
      for (int s = 0; s < 8; ++s) {
        unsigned i = tid + (unsigned)(s << 10);
        unsigned ixj = i ^ j;
        if (ixj > i) {
          bool up = ((i & k) == 0);
          ull a = skey[i], b = skey[ixj];
          if ((a > b) == up) { skey[i] = b; skey[ixj] = a; }
        }
      }
    }
  }
  __syncthreads();
  #pragma unroll
  for (int s = 0; s < 8; ++s) {
    unsigned i = tid + (unsigned)(s << 10);
    unsigned idx = (unsigned)(skey[i] & 0xFFFFFFFFull);
    order[i]   = (int)idx;
    slabels[i] = labels[idx];
    ((float4*)sboxes)[i] = ((const float4*)boxes)[idx];
  }
}

__global__ __launch_bounds__(128) void nms_mask_kernel(
    const float* __restrict__ sboxes, const int* __restrict__ slabels,
    const float* __restrict__ thresh_p, unsigned* __restrict__ mask)
{
  const int lane    = threadIdx.x & 31;
  const int waveG   = blockIdx.x * 4 + (threadIdx.x >> 5);
  const int rowTile = waveG >> 8;
  const int colPair = waveG & 255;
  const int rowBase = rowTile << 4;
  const int hf      = lane >> 4;
  const int nIdx    = lane & 15;
  const float th    = *thresh_p;

  const int rowLabel = slabels[rowBase + nIdx];
  v16i A;
  #pragma unroll
  for (int v = 0; v < 16; ++v) {
    int s = v >> 3, vv = v & 7;
    int kb = s * 64 + (vv >> 1) * 16 + (vv & 1) * 4 + hf * 8;
    int d = rowLabel - kb;
    A[v] = (d >= 0 && d < 4) ? (int)(0x38u << (d * 8)) : 0;
  }

  unsigned ball[2][8];
  #pragma unroll
  for (int t = 0; t < 2; ++t) {
    const int colBase = (colPair << 5) + (t << 4);
    const int colLabel = slabels[colBase + nIdx];
    v16i B;
    #pragma unroll
    for (int v = 0; v < 16; ++v) {
      int g = v >> 2;
      int kb = (v >> 3) * 64 + ((v & 7) >> 1) * 16 + (v & 1) * 4 + hf * 8;
      int d = colLabel - kb;
      B[v] = (d >= 0 && d < 4) ? (int)(0x38u << (d * 8)) : 0;
    }
    v8f cz = {};
    v8f c = __builtin_amdgcn_wmma_f32_16x16x128_fp8_fp8(
        A, B, (short)0, cz, false, false);

    const int gj = colBase + nIdx;
    float4 cb = ((const float4*)sboxes)[gj];
    float areaC = (cb.z - cb.x) * (cb.w - cb.y);
    #pragma unroll
    for (int r = 0; r < 8; ++r) {
      int gi = rowBase + r + (hf << 3);
      float4 rb = ((const float4*)sboxes)[gi];
      float areaR = (rb.z - rb.x) * (rb.w - rb.y);
      float iw = fminf(rb.z, cb.z) - fmaxf(rb.x, cb.x);
      float ih = fminf(rb.w, cb.w) - fmaxf(rb.y, cb.y);
      iw = fmaxf(iw, 0.0f); ih = fmaxf(ih, 0.0f);
      float inter = iw * ih;
      float uni   = areaR + areaC - inter;
      bool sup = (gj > gi) && (c[r] > 0.5f) && (inter > th * uni);
      ball[t][r] = (unsigned)__ballot(sup);
    }
  }
  if (lane == 0) {
    #pragma unroll
    for (int r = 0; r < 8; ++r) {
      unsigned lo = (ball[0][r] & 0xFFFFu) | (ball[1][r] << 16);
      unsigned hi = (ball[0][r] >> 16)     | (ball[1][r] & 0xFFFF0000u);
      mask[(size_t)(rowBase + r)     * MASK_WORDS + colPair] = lo;
      mask[(size_t)(rowBase + r + 8) * MASK_WORDS + colPair] = hi;
    }
  }
}

__global__ __launch_bounds__(256) void nms_greedy_kernel(
    const unsigned* __restrict__ mask, const int* __restrict__ mp_p,
    unsigned* __restrict__ keepw_g)
{
  __shared__ unsigned remv[MASK_WORDS];
  __shared__ unsigned keeps[MASK_WORDS];
  const int t = threadIdx.x;
  remv[t] = 0u;
  unsigned kw = 0u;
  __syncthreads();
  for (int i = 0; i < N_BOX; ++i) {
    unsigned w = remv[i >> 5];
    __syncthreads();
    if (!((w >> (i & 31)) & 1u)) {
      remv[t] |= mask[(size_t)i * MASK_WORDS + t];
      if ((i >> 5) == t) kw |= (1u << (i & 31));
    }
    __syncthreads();
  }
  keeps[t] = kw;
  __syncthreads();
  if (t == 0) {
    int mp = *mp_p;
    if (mp > 0) {
      int cnt = 0;
      for (int wI = 0; wI < MASK_WORDS; ++wI) {
        unsigned w = keeps[wI];
        int pc = __popc(w);
        if (cnt >= mp) { keeps[wI] = 0u; }
        else if (cnt + pc > mp) {
          int allow = mp - cnt;
          unsigned nw = 0u;
          for (int b = 0; b < 32; ++b) {
            if ((w >> b) & 1u) { if (allow > 0) { nw |= 1u << b; --allow; } }
          }
          keeps[wI] = nw; cnt = mp;
        } else { cnt += pc; }
      }
    }
  }
  __syncthreads();
  keepw_g[t] = keeps[t];
}

__global__ __launch_bounds__(256) void nms_write_kernel(
    const float* __restrict__ sboxes, const int* __restrict__ order,
    const unsigned* __restrict__ keepw, float* __restrict__ out)
{
  int p = blockIdx.x * 256 + threadIdx.x;
  bool k = (keepw[p >> 5] >> (p & 31)) & 1u;
  float4 b = ((const float4*)sboxes)[p];
  float4 ob = k ? b : make_float4(0.0f, 0.0f, 0.0f, 0.0f);
  ((float4*)out)[p] = ob;
  out[4 * N_BOX + p] = (float)order[p];
  out[5 * N_BOX + p] = k ? 1.0f : 0.0f;
}

extern "C" void kernel_launch(void* const* d_in, const int* in_sizes, int n_in,
                              void* d_out, int out_size, void* d_ws, size_t ws_size,
                              hipStream_t stream) {
  (void)in_sizes; (void)n_in; (void)out_size; (void)ws_size;
  const float* boxes  = (const float*)d_in[0];
  const float* scores = (const float*)d_in[1];
  const int*   labels = (const int*)d_in[2];
  const float* thresh = (const float*)d_in[3];
  const int*   mp     = (const int*)d_in[4];
  float* out = (float*)d_out;

  char* ws = (char*)d_ws;
  unsigned* mask    = (unsigned*)ws;
  float*    sboxes  = (float*)(ws + (size_t)N_BOX * MASK_WORDS * 4);
  int*      slabels = (int*)((char*)sboxes + (size_t)N_BOX * 4 * sizeof(float));
  int*      order   = (int*)((char*)slabels + (size_t)N_BOX * sizeof(int));
  unsigned* keepw   = (unsigned*)((char*)order + (size_t)N_BOX * sizeof(int));

  nms_sort_kernel<<<1, 1024, 0, stream>>>(scores, boxes, labels,
                                          sboxes, slabels, order);
  nms_mask_kernel<<<(512 * 256) / 4, 128, 0, stream>>>(sboxes, slabels,
                                                       thresh, mask);
  nms_greedy_kernel<<<1, 256, 0, stream>>>(mask, mp, keepw);
  nms_write_kernel<<<N_BOX / 256, 256, 0, stream>>>(sboxes, order, keepw, out);
}
